// GTN_83227876262556
// MI455X (gfx1250) — matrix operand order repaired and checked by reading, not compiled, not run on hardware
//
#include <hip/hip_runtime.h>
#include <hip/hip_bf16.h>

// ---------------------------------------------------------------------------
// GTN forward for MI455X (gfx1250, wave32, WMMA).
// Heavy work: two C=4-batched 2048^3 GEMMs + one 2048x2048x128 GEMM, all done
// with v_wmma_f32_16x16x32_f16 (f16 operands, f32 accumulate), software-
// pipelined double-buffered LDS staging, async global->LDS (CDNA5 path).
// ---------------------------------------------------------------------------

typedef __attribute__((ext_vector_type(16))) _Float16 v16h;
typedef __attribute__((ext_vector_type(8)))  _Float16 v8h;
typedef __attribute__((ext_vector_type(8)))  float    v8f;
typedef __attribute__((ext_vector_type(4)))  int      v4i;

// Explicit global-address-space types: keep staging loads / epilogue stores on
// the GLOBAL_* path (flat_* would also tick DScnt and use the LDS datapath).
typedef const __attribute__((address_space(1))) v8h   g_v8h;
typedef __attribute__((address_space(1))) float       g_f32;

static constexpr int       Nn   = 2048;
static constexpr int       Ee   = 5;
static constexpr int       Cc   = 4;
static constexpr int       WIN  = 256;
static constexpr int       WOUT = 128;
static constexpr long long NN   = (long long)Nn * Nn;

// ---- CDNA5 async global->LDS availability probe -----------------------------
#if defined(__has_builtin)
#if __has_builtin(__builtin_amdgcn_global_load_async_to_lds_b128)
#define GTN_HAS_ASYNC 1
#endif
#endif
#ifndef GTN_HAS_ASYNC
#define GTN_HAS_ASYNC 0
#endif

#if GTN_HAS_ASYNC
typedef __attribute__((address_space(1))) v4i as1_v4i;  // global
typedef __attribute__((address_space(3))) v4i as3_v4i;  // LDS
__device__ __forceinline__ void gtn_async_b128(const _Float16* g, _Float16* l) {
  __builtin_amdgcn_global_load_async_to_lds_b128(
      (as1_v4i*)g, (as3_v4i*)l, 0, 0);
}
__device__ __forceinline__ void gtn_wait_async() {
#if __has_builtin(__builtin_amdgcn_s_wait_asynccnt)
  __builtin_amdgcn_s_wait_asynccnt(0);
#else
  asm volatile("s_wait_asynccnt 0x0" ::: "memory");
#endif
}
#endif

// ---------------------------------------------------------------------------
// softmax(w, axis=E) mix of adjacency stack: Q[c] = sum_e f[c,e] * A[e]
// ---------------------------------------------------------------------------
__global__ __launch_bounds__(256) void gtconv_kernel(
    const float* __restrict__ A, const float* __restrict__ wA,
    const float* __restrict__ wB, _Float16* __restrict__ Qa,
    _Float16* __restrict__ Qb, int nOut) {
  __shared__ float fA[Cc * Ee];
  __shared__ float fB[Cc * Ee];
  if (threadIdx.x == 0) {
    for (int c = 0; c < Cc; ++c) {
      float mx = wA[c * Ee];
      for (int e = 1; e < Ee; ++e) mx = fmaxf(mx, wA[c * Ee + e]);
      float s = 0.f, ex[Ee];
      for (int e = 0; e < Ee; ++e) { ex[e] = __expf(wA[c * Ee + e] - mx); s += ex[e]; }
      for (int e = 0; e < Ee; ++e) fA[c * Ee + e] = ex[e] / s;
    }
    if (nOut > 1) {
      for (int c = 0; c < Cc; ++c) {
        float mx = wB[c * Ee];
        for (int e = 1; e < Ee; ++e) mx = fmaxf(mx, wB[c * Ee + e]);
        float s = 0.f, ex[Ee];
        for (int e = 0; e < Ee; ++e) { ex[e] = __expf(wB[c * Ee + e] - mx); s += ex[e]; }
        for (int e = 0; e < Ee; ++e) fB[c * Ee + e] = ex[e] / s;
      }
    }
  }
  __syncthreads();
  long long idx = (long long)blockIdx.x * 256 + threadIdx.x;  // < NN
  float a[Ee];
#pragma unroll
  for (int e = 0; e < Ee; ++e) a[e] = A[(long long)e * NN + idx];
#pragma unroll
  for (int c = 0; c < Cc; ++c) {
    float s = 0.f;
#pragma unroll
    for (int e = 0; e < Ee; ++e) s += fA[c * Ee + e] * a[e];
    Qa[(long long)c * NN + idx] = (_Float16)s;
  }
  if (nOut > 1) {
#pragma unroll
    for (int c = 0; c < Cc; ++c) {
      float s = 0.f;
#pragma unroll
      for (int e = 0; e < Ee; ++e) s += fB[c * Ee + e] * a[e];
      Qb[(long long)c * NN + idx] = (_Float16)s;
    }
  }
}

// ---------------------------------------------------------------------------
// Generic batched WMMA GEMM: C[m,n] = sum_k A[m,k]*B[k,n]  (f16 in, f32 out).
// 256 threads = 8 waves; tile 128x128; K-step 32; double-buffered LDS with a
// one-iteration software pipeline (loads for tile i+1 issued before wmma of
// tile i). A tiles go global->LDS via CDNA5 async-to-LDS when available.
// TRANSA: A supplied as [K][M] row-major. BIAS_RELU: out = relu(acc+bias[n]).
// ---------------------------------------------------------------------------
template <bool TRANSA, bool BIAS_RELU>
__global__ __launch_bounds__(256) void wmma_gemm_kernel(
    const _Float16* __restrict__ Ag, long long strideA, int lda,
    const _Float16* __restrict__ Bg, long long strideB, int ldb,
    float* __restrict__ Cg, long long strideC, int ldc,
    const float* __restrict__ bias, int K) {
  constexpr int KT = 32;
  constexpr int LS = 48;  // LDS row stride in f16 (96B, 16B aligned)
  __shared__ __align__(16) _Float16 As[2][128 * LS];
  __shared__ __align__(16) _Float16 Bt[2][128 * LS];

  const int t    = threadIdx.x;
  const int lane = t & 31;
  const int wid  = t >> 5;
  const int wm   = wid & 3;   // *32 rows
  const int wn   = wid >> 2;  // *64 cols
  const int half = lane >> 4;
  const int l16  = lane & 15;
  const int bm   = blockIdx.x * 128;
  const int bn   = blockIdx.y * 128;

  const _Float16* Ab = Ag + (long long)blockIdx.z * strideA;
  const _Float16* Bb = Bg + (long long)blockIdx.z * strideB;
  float*          Cb = Cg + (long long)blockIdx.z * strideC;

  // ---- per-thread staging coordinates (constant), pointers bumped per tile --
  int aRow[2] = {0, 0}, aCol[2] = {0, 0};  // !TRANSA
  int aK[2] = {0, 0}, aI0[2] = {0, 0};     // TRANSA
  int bK[2], bN0[2];
  const _Float16* aPtr[2];
  const _Float16* bPtr[2];
#pragma unroll
  for (int r = 0; r < 2; ++r) {
    int idx = t + r * 256;  // 0..511
    if (!TRANSA) {
      aRow[r] = idx >> 2;           // 0..127
      aCol[r] = (idx & 3) << 3;     // 0,8,16,24
      aPtr[r] = Ab + (long long)(bm + aRow[r]) * lda + aCol[r];
    } else {
      aK[r]  = idx >> 4;            // 0..31
      aI0[r] = (idx & 15) << 3;     // 0..120
      aPtr[r] = Ab + (long long)aK[r] * lda + bm + aI0[r];
    }
    bK[r]  = idx >> 4;
    bN0[r] = (idx & 15) << 3;
    bPtr[r] = Bb + (long long)bK[r] * ldb + bn + bN0[r];
  }
  const long long aStep = TRANSA ? (long long)KT * lda : (long long)KT;
  const long long bStep = (long long)KT * ldb;

  v8f acc[2][4];
#pragma unroll
  for (int s = 0; s < 2; ++s)
#pragma unroll
    for (int q = 0; q < 4; ++q)
#pragma unroll
      for (int i = 0; i < 8; ++i) acc[s][q][i] = 0.f;

  union V16 { v16h v; v8h h[2]; };

  const int nT = K / KT;
  v8h aReg[2], bReg[2];

  // ---- prologue: issue loads for tile 0 ----
#pragma unroll
  for (int r = 0; r < 2; ++r) { bReg[r] = *(g_v8h*)bPtr[r]; bPtr[r] += bStep; }
  if (TRANSA) {
#pragma unroll
    for (int r = 0; r < 2; ++r) { aReg[r] = *(g_v8h*)aPtr[r]; aPtr[r] += aStep; }
  } else {
#if GTN_HAS_ASYNC
#pragma unroll
    for (int r = 0; r < 2; ++r) {
      gtn_async_b128(aPtr[r], &As[0][aRow[r] * LS + aCol[r]]);
      aPtr[r] += aStep;
    }
#else
#pragma unroll
    for (int r = 0; r < 2; ++r) { aReg[r] = *(g_v8h*)aPtr[r]; aPtr[r] += aStep; }
#endif
  }

  for (int i = 0; i < nT; ++i) {
    const int cur = i & 1;
    _Float16* AsC = As[cur];
    _Float16* BtC = Bt[cur];

    // ---- commit register-staged tile i into LDS ----
#pragma unroll
    for (int r = 0; r < 2; ++r)
#pragma unroll
      for (int j = 0; j < 8; ++j) BtC[(bN0[r] + j) * LS + bK[r]] = bReg[r][j];
    if (TRANSA) {
#pragma unroll
      for (int r = 0; r < 2; ++r)
#pragma unroll
        for (int j = 0; j < 8; ++j) AsC[(aI0[r] + j) * LS + aK[r]] = aReg[r][j];
    } else {
#if !GTN_HAS_ASYNC
#pragma unroll
      for (int r = 0; r < 2; ++r) *(v8h*)&AsC[aRow[r] * LS + aCol[r]] = aReg[r];
#endif
    }
#if GTN_HAS_ASYNC
    if (!TRANSA) gtn_wait_async();  // tile i async A landed in As[cur]
#endif
    __syncthreads();

    // ---- issue loads for tile i+1 (overlap with wmma below) ----
    if (i + 1 < nT) {
#pragma unroll
      for (int r = 0; r < 2; ++r) { bReg[r] = *(g_v8h*)bPtr[r]; bPtr[r] += bStep; }
      if (TRANSA) {
#pragma unroll
        for (int r = 0; r < 2; ++r) { aReg[r] = *(g_v8h*)aPtr[r]; aPtr[r] += aStep; }
      } else {
#if GTN_HAS_ASYNC
#pragma unroll
        for (int r = 0; r < 2; ++r) {
          gtn_async_b128(aPtr[r], &As[cur ^ 1][aRow[r] * LS + aCol[r]]);
          aPtr[r] += aStep;
        }
#else
#pragma unroll
        for (int r = 0; r < 2; ++r) {
          aReg[r] = *(g_v8h*)aPtr[r];
          aPtr[r] += aStep;
        }
        __builtin_prefetch(bPtr[0], 0, 1);  // gfx1250 global_prefetch_b8
#endif
      }
    }

    // ---- fragments (ISA 16-bit A/B VGPR layouts) + 2x4 wmma ----
    v16h af[2];
#pragma unroll
    for (int s = 0; s < 2; ++s) {
      int row = wm * 32 + s * 16 + l16;  // m = lane%16
      V16 u;
      u.h[0] = *(const v8h*)&AsC[row * LS + half * 8];       // K[half*8 .. +8)
      u.h[1] = *(const v8h*)&AsC[row * LS + 16 + half * 8];  // K[16+half*8 .. +8)
      af[s] = u.v;
    }
    v16h bf[4];
#pragma unroll
    for (int q = 0; q < 4; ++q) {
      int nr = wn * 64 + q * 16 + l16;   // n = lane%16
      V16 u;
      u.h[0] = *(const v8h*)&BtC[nr * LS + half * 8];        // K[half*8 .. +8)
      u.h[1] = *(const v8h*)&BtC[nr * LS + 16 + half * 8];
      bf[q] = u.v;
    }
#pragma unroll
    for (int s = 0; s < 2; ++s)
#pragma unroll
      for (int q = 0; q < 4; ++q)
        acc[s][q] = __builtin_amdgcn_wmma_f32_16x16x32_f16(
            false, af[s], false, bf[q], (short)0, acc[s][q], false, false);
    // no trailing barrier: next iteration writes the other LDS buffer, and its
    // top-of-loop barrier orders buffer reuse two iterations apart.
  }

  // ---- epilogue: C/D layout lane%16 = n, VGPR r -> m = r + (lane/16)*8 ----
#pragma unroll
  for (int s = 0; s < 2; ++s) {
#pragma unroll
    for (int q = 0; q < 4; ++q) {
      int col = bn + wn * 64 + q * 16 + l16;
      float bv = BIAS_RELU ? bias[col] : 0.f;
#pragma unroll
      for (int r = 0; r < 8; ++r) {
        int row = bm + wm * 32 + s * 16 + half * 8 + r;
        float v = acc[s][q][r];
        if (BIAS_RELU) v = fmaxf(v + bv, 0.f);
        *(g_f32*)&Cb[(long long)row * ldc + col] = v;
      }
    }
  }
}

// ---------------------------------------------------------------------------
// deg[c,j] = sum_{r!=j} H[c,r,j];  dinv = deg!=0 ? 1/deg : 0
// ---------------------------------------------------------------------------
__global__ __launch_bounds__(256) void colsum_dinv_kernel(
    const float* __restrict__ H, float* __restrict__ dinv) {
  int j = blockIdx.x * 256 + threadIdx.x;
  int c = blockIdx.y;
  const float* Hc = H + (long long)c * NN;
  float s = 0.f;
  for (int r = 0; r < Nn; ++r) {
    float x = Hc[(long long)r * Nn + j];
    s += (r == j) ? 0.f : x;
  }
  dinv[c * Nn + j] = (s != 0.f) ? 1.f / s : 0.f;
}

// degHat[c,j] = 1 + sum_{r!=j} H2[c,r,j]*dinv2[c,r];  dis = rsqrt(degHat)
__global__ __launch_bounds__(256) void colsum_dis_kernel(
    const float* __restrict__ H2, const float* __restrict__ dinv2,
    float* __restrict__ dis) {
  int j = blockIdx.x * 256 + threadIdx.x;
  int c = blockIdx.y;
  const float* Hc = H2 + (long long)c * NN;
  const float* dv = dinv2 + c * Nn;
  float s = 0.f;
  for (int r = 0; r < Nn; ++r) {
    float x = Hc[(long long)r * Nn + j] * dv[r];
    s += (r == j) ? 0.f : x;
  }
  float d = 1.f + s;
  dis[c * Nn + j] = (d > 0.f) ? rsqrtf(d) : 0.f;
}

// H_norm[c,r,m] = (r==m ? 0 : H[c,r,m]*dinv[c,r])  -> f16
__global__ __launch_bounds__(256) void scale_f16_kernel(
    const float* __restrict__ H, const float* __restrict__ dinv,
    _Float16* __restrict__ out) {
  long long idx = (long long)blockIdx.x * 256 + threadIdx.x;  // < NN
  int c = blockIdx.y;
  int r = (int)(idx >> 11);
  int m = (int)(idx & 2047);
  float v = (r == m) ? 0.f : H[(long long)c * NN + idx] * dinv[c * Nn + r];
  out[(long long)c * NN + idx] = (_Float16)v;
}

// norm[c,r,m] = dis[c,r]*Hat[c,r,m]*dis[c,m]; Hat = (r==m?1:H2*dinv2[c,r]) -> f16
__global__ __launch_bounds__(256) void norm_f16_kernel(
    const float* __restrict__ H2, const float* __restrict__ dinv2,
    const float* __restrict__ dis, _Float16* __restrict__ out) {
  long long idx = (long long)blockIdx.x * 256 + threadIdx.x;  // < NN
  int c = blockIdx.y;
  int r = (int)(idx >> 11);
  int m = (int)(idx & 2047);
  float hat = (r == m) ? 1.f : H2[(long long)c * NN + idx] * dinv2[c * Nn + r];
  out[(long long)c * NN + idx] = (_Float16)(dis[c * Nn + r] * hat * dis[c * Nn + m]);
}

// XW[n,f] = (X @ gcn_w)[n,f] -> f16   (small: 2048x256x128)
__global__ __launch_bounds__(128) void xw_kernel(
    const float* __restrict__ X, const float* __restrict__ W,
    _Float16* __restrict__ XW) {
  int f = threadIdx.x;
  int n = blockIdx.x;
  float s = 0.f;
  for (int i = 0; i < WIN; ++i) s += X[n * WIN + i] * W[i * WOUT + f];
  XW[(long long)n * WOUT + f] = (_Float16)s;
}

// h[n,o] = relu( sum_{k,f} G[k,n,f]*W[(k*F+f),o] + b[o] )
__global__ __launch_bounds__(128) void lin1_kernel(
    const float* __restrict__ G, const float* __restrict__ W,
    const float* __restrict__ b, float* __restrict__ h) {
  int o = threadIdx.x;
  int n = blockIdx.x;
  float s = b[o];
  for (int k = 0; k < Cc; ++k)
    for (int f = 0; f < WOUT; ++f)
      s += G[((long long)k * Nn + n) * WOUT + f] * W[(k * WOUT + f) * WOUT + o];
  h[(long long)n * WOUT + o] = fmaxf(s, 0.f);
}

// out[n,o] = sum_j h[n,j]*W[j,o] + b[o]
__global__ __launch_bounds__(128) void lin2_kernel(
    const float* __restrict__ h, const float* __restrict__ W,
    const float* __restrict__ b, float* __restrict__ out) {
  int o = threadIdx.x;
  int n = blockIdx.x;
  float s = b[o];
  for (int j = 0; j < WOUT; ++j) s += h[(long long)n * WOUT + j] * W[j * WOUT + o];
  out[(long long)n * WOUT + o] = s;
}

// ---------------------------------------------------------------------------
extern "C" void kernel_launch(void* const* d_in, const int* in_sizes, int n_in,
                              void* d_out, int out_size, void* d_ws, size_t ws_size,
                              hipStream_t stream) {
  const float* A      = (const float*)d_in[0];
  const float* X      = (const float*)d_in[1];
  const float* w0a    = (const float*)d_in[2];
  const float* w0b    = (const float*)d_in[3];
  const float* w1     = (const float*)d_in[4];
  const float* gcn_w  = (const float*)d_in[5];
  const float* gcn_b  = (const float*)d_in[6];
  const float* lin1_w = (const float*)d_in[7];
  const float* lin1_b = (const float*)d_in[8];
  const float* lin2_w = (const float*)d_in[9];
  const float* lin2_b = (const float*)d_in[10];
  float* out = (float*)d_out;

  // Workspace layout with buffer reuse (peak ~140 MB):
  //   slot0 (f16, C*NN): Qa  -> H1_norm -> norm
  //   slot1 (f16, C*NN): Qb  -> Q1
  //   slot2 (f32, C*NN): H1  -> H2
  //   small region: dinv1, dinv2, dis, XW(f16), G(f32), h(f32)
  char* ws = (char*)d_ws;
  const size_t SLOT16 = (size_t)Cc * NN * sizeof(_Float16);  // 33,554,432
  const size_t SLOT32 = (size_t)Cc * NN * sizeof(float);     // 67,108,864
  _Float16* slot0 = (_Float16*)(ws);
  _Float16* slot1 = (_Float16*)(ws + SLOT16);
  float*    slot2 = (float*)(ws + 2 * SLOT16);
  char* small = ws + 2 * SLOT16 + SLOT32;
  float*    dinv1 = (float*)(small);
  float*    dinv2 = (float*)(small + 32768);
  float*    dis   = (float*)(small + 65536);
  _Float16* XW    = (_Float16*)(small + 98304);
  float*    G     = (float*)(small + 98304 + 524288);
  float*    h     = (float*)(small + 98304 + 524288 + 4194304);

  const dim3 blk256(256), blk128(128);
  const dim3 gridNN((unsigned)(NN / 256), Cc);
  const dim3 gridN(Nn / 256, Cc);
  const dim3 gridElem((unsigned)(NN / 256));
  const dim3 gGemmBig(Nn / 128, Nn / 128, Cc);
  const dim3 gGemmGcn(Nn / 128, WOUT / 128, Cc);

  // 1) Qa,Qb = softmax-mixed adjacencies (f16)
  gtconv_kernel<<<gridElem, blk256, 0, stream>>>(A, w0a, w0b, slot0, slot1, 2);
  // 2) H1 = Qa @ Qb   (batched 2048^3, WMMA)
  wmma_gemm_kernel<false, false><<<gGemmBig, blk256, 0, stream>>>(
      slot0, NN, Nn, slot1, NN, Nn, slot2, NN, Nn, nullptr, Nn);
  // 3) column degrees (diag excluded) -> dinv1
  colsum_dinv_kernel<<<gridN, blk256, 0, stream>>>(slot2, dinv1);
  // 4) H1_norm (f16) -> slot0
  scale_f16_kernel<<<gridNN, blk256, 0, stream>>>(slot2, dinv1, slot0);
  // 5) Q1 = softmax-mixed adjacency for layer 1 -> slot1
  gtconv_kernel<<<gridElem, blk256, 0, stream>>>(A, w1, w1, slot1, nullptr, 1);
  // 6) H2 = H1_norm @ Q1 -> slot2
  wmma_gemm_kernel<false, false><<<gGemmBig, blk256, 0, stream>>>(
      slot0, NN, Nn, slot1, NN, Nn, slot2, NN, Nn, nullptr, Nn);
  // 7) dinv2 from H2
  colsum_dinv_kernel<<<gridN, blk256, 0, stream>>>(slot2, dinv2);
  // 8) dis = rsqrt(1 + colsum(H2_norm))
  colsum_dis_kernel<<<gridN, blk256, 0, stream>>>(slot2, dinv2, dis);
  // 9) norm (f16) -> slot0
  norm_f16_kernel<<<gridNN, blk256, 0, stream>>>(slot2, dinv2, dis, slot0);
  // 10) XW = X @ gcn_w (f16)
  xw_kernel<<<dim3(Nn), blk128, 0, stream>>>(X, gcn_w, XW);
  // 11) G[k] = relu(norm[k]^T @ XW + gcn_b)   (TRANSA WMMA, fused epilogue)
  wmma_gemm_kernel<true, true><<<gGemmGcn, blk256, 0, stream>>>(
      slot0, NN, Nn, XW, 0, WOUT, G, (long long)Nn * WOUT, WOUT, gcn_b, Nn);
  // 12) lin1 + relu
  lin1_kernel<<<dim3(Nn), blk128, 0, stream>>>(G, lin1_w, lin1_b, h);
  // 13) lin2 -> output
  lin2_kernel<<<dim3(Nn), blk128, 0, stream>>>(h, lin2_w, lin2_b, out);

  (void)in_sizes; (void)n_in; (void)out_size; (void)ws_size;
}
